// MHA_40458591929230
// MI455X (gfx1250) — matrix operand order repaired and checked by reading, not compiled, not run on hardware
//
#include <hip/hip_runtime.h>

// ---------------- problem constants ----------------
#define EMBED 1024
#define NHEAD 16
#define HDIM  64
#define LSEQ  2048
#define BSZ   2
#define SSEQ  2048
#define MTOK  (LSEQ * BSZ) // 4096 tokens (rows of all dense GEMMs)

typedef _Float16 v16h __attribute__((ext_vector_type(16)));
typedef _Float16 v8h  __attribute__((ext_vector_type(8)));
typedef float    v8f  __attribute__((ext_vector_type(8)));

// ---------------- WMMA helpers ----------------
__device__ __forceinline__ v8f wmma_f16(v16h a, v16h b, v8f c) {
    // D = A(16x32 f16) * B(32x16 f16) + C(16x16 f32)
    return __builtin_amdgcn_wmma_f32_16x16x32_f16(
        false, a, false, b, (short)0, c, false, false);
}

// A-matrix fragment (16x32, MxK). Layout: lanes 0-15 rows M=0..15 with
// K = 0..15; lanes 16-31 same rows with K = 16..31.
__device__ __forceinline__ v16h load_frag_a(const _Float16* base, int ldk, int lane) {
    int r    = lane & 15;
    int koff = (lane >> 4) << 4; // 0 or 16
    const _Float16* p = base + (size_t)r * ldk + koff;
    union { v16h v; v8h h[2]; } u;
    u.h[0] = *(const v8h*)(p);       // K koff .. koff+7
    u.h[1] = *(const v8h*)(p + 8);   // K koff+8 .. koff+15
    return u.v;
}

// B-matrix fragment (32x16, KxN). Lanes 0-15 hold column N=lane with K=0..15;
// lanes 16-31 hold column N=lane-16 with K=16..31. Memory is column-major
// along K with leading dim ldk.
__device__ __forceinline__ v16h load_frag_b(const _Float16* base, int ldk, int lane) {
    int c    = lane & 15;
    int koff = (lane >> 4) << 4; // 0 or 16
    const _Float16* p = base + (size_t)c * ldk + koff;
    union { v16h v; v8h h[2]; } u;
    u.h[0] = *(const v8h*)(p);      // K koff .. koff+7
    u.h[1] = *(const v8h*)(p + 8);  // K koff+8 .. koff+15
    return u.v;
}

// max/sum across the 16-lane group that owns one set of 8 rows
__device__ __forceinline__ float group_max(float v) {
    #pragma unroll
    for (int m = 1; m <= 8; m <<= 1) v = fmaxf(v, __shfl_xor(v, m, 32));
    return v;
}
__device__ __forceinline__ float group_sum(float v) {
    #pragma unroll
    for (int m = 1; m <= 8; m <<= 1) v += __shfl_xor(v, m, 32);
    return v;
}

// ---------------- CDNA5 async copy (global -> LDS, ASYNCcnt-tracked) ----------
// Copies one 8 KB K-chunk (64 rows x 64 f16) with 4 async b128 loads per thread
// (128 threads). VDST operand = low 32 bits of the flat shared-pointer (= LDS
// byte offset); VADDR = 64-bit global address.
__device__ __forceinline__ void stage_k_chunk(const _Float16* g, _Float16* l, int tid) {
#if defined(__gfx1250__)
    unsigned           lbase = (unsigned)(unsigned long long)l;
    unsigned long long gbase = (unsigned long long)g;
    #pragma unroll
    for (int i = 0; i < 4; i++) {
        unsigned           loff = lbase + (unsigned)((i * 128 + tid) * 16);
        unsigned long long ga   = gbase + (unsigned long long)((i * 128 + tid) * 16);
        asm volatile("global_load_async_to_lds_b128 %0, %1, off"
                     :: "v"(loff), "v"(ga) : "memory");
    }
#else
    v8h* d = (v8h*)l; const v8h* s = (const v8h*)g;
    for (int i = 0; i < 4; i++) d[i * 128 + tid] = s[i * 128 + tid];
#endif
}
__device__ __forceinline__ void async_wait_le4() {
#if defined(__gfx1250__)
    asm volatile("s_wait_asynccnt 0x4" ::: "memory");
#endif
}
__device__ __forceinline__ void async_wait_0() {
#if defined(__gfx1250__)
    asm volatile("s_wait_asynccnt 0x0" ::: "memory");
#endif
}

// ---------------- fp32 -> fp16 conversion ----------------
__global__ __launch_bounds__(256) void cvt4_kernel(const float4* __restrict__ src,
                                                   _Float16* __restrict__ dst, int n4) {
    int i = blockIdx.x * blockDim.x + threadIdx.x;
    if (i < n4) {
        float4 f = src[i];
        _Float16* d = dst + (size_t)i * 4;
        d[0] = (_Float16)f.x; d[1] = (_Float16)f.y;
        d[2] = (_Float16)f.z; d[3] = (_Float16)f.w;
    }
}

// ---------------- GEMM: Y = X(4096x1024) * W^T + bias ----------------
// Each wave: 32(M) x 64(N) tile (2 A-frags x 4 B-frags = 8 WMMAs / K-step).
// Block: 8 waves (2 x 32M, 4 x 64N) = 64M x 256N. Grid: (EMBED/256, MTOK/64).
__global__ __launch_bounds__(256) void gemm_head_kernel(
    const _Float16* __restrict__ X, const _Float16* __restrict__ Wt,
    const float* __restrict__ bias, _Float16* __restrict__ dst) {
    int tid = threadIdx.x, lane = tid & 31, w = tid >> 5;
    int m0 = blockIdx.y * 64 + (w & 1) * 32;
    int n0 = blockIdx.x * 256 + (w >> 1) * 64;
    v8f acc[2][4] = {};
    for (int kt = 0; kt < EMBED; kt += 32) {
        __builtin_prefetch(X + (size_t)m0 * EMBED + kt + 128, 0, 1);
        v16h a0 = load_frag_a(X + (size_t)m0 * EMBED + kt, EMBED, lane);
        v16h a1 = load_frag_a(X + (size_t)(m0 + 16) * EMBED + kt, EMBED, lane);
        #pragma unroll
        for (int j = 0; j < 4; j++) {
            v16h b = load_frag_b(Wt + (size_t)(n0 + j * 16) * EMBED + kt, EMBED, lane);
            acc[0][j] = wmma_f16(a0, b, acc[0][j]);
            acc[1][j] = wmma_f16(a1, b, acc[1][j]);
        }
    }
    int mb = (lane >> 4) << 3;
    #pragma unroll
    for (int mi = 0; mi < 2; mi++) {
        #pragma unroll
        for (int j = 0; j < 4; j++) {
            #pragma unroll
            for (int r = 0; r < 8; r++) {
                int m = m0 + mi * 16 + r + mb;     // token index = l*B + b
                int n = n0 + j * 16 + (lane & 15);
                int hh = n >> 6, dd = n & 63;
                int bb = m & (BSZ - 1), ll = m >> 1;
                float v = acc[mi][j][r] + bias[n];
                dst[((size_t)(bb * NHEAD + hh) * LSEQ + ll) * HDIM + dd] = (_Float16)v;
            }
        }
    }
}

__global__ __launch_bounds__(256) void gemm_out_kernel(
    const _Float16* __restrict__ X, const _Float16* __restrict__ Wt,
    const float* __restrict__ bias, float* __restrict__ out) {
    int tid = threadIdx.x, lane = tid & 31, w = tid >> 5;
    int m0 = blockIdx.y * 64 + (w & 1) * 32;
    int n0 = blockIdx.x * 256 + (w >> 1) * 64;
    v8f acc[2][4] = {};
    for (int kt = 0; kt < EMBED; kt += 32) {
        __builtin_prefetch(X + (size_t)m0 * EMBED + kt + 128, 0, 1);
        v16h a0 = load_frag_a(X + (size_t)m0 * EMBED + kt, EMBED, lane);
        v16h a1 = load_frag_a(X + (size_t)(m0 + 16) * EMBED + kt, EMBED, lane);
        #pragma unroll
        for (int j = 0; j < 4; j++) {
            v16h b = load_frag_b(Wt + (size_t)(n0 + j * 16) * EMBED + kt, EMBED, lane);
            acc[0][j] = wmma_f16(a0, b, acc[0][j]);
            acc[1][j] = wmma_f16(a1, b, acc[1][j]);
        }
    }
    int mb = (lane >> 4) << 3;
    #pragma unroll
    for (int mi = 0; mi < 2; mi++) {
        #pragma unroll
        for (int j = 0; j < 4; j++) {
            #pragma unroll
            for (int r = 0; r < 8; r++) {
                int m = m0 + mi * 16 + r + mb;
                int n = n0 + j * 16 + (lane & 15);
                out[(size_t)m * EMBED + n] = acc[mi][j][r] + bias[n];
            }
        }
    }
}

// ---------------- flash attention ----------------
// Grid: (LSEQ/64, B*H). Block: 128 threads = 4 waves; wave owns 16 Q rows.
// K chunks are double-buffered in LDS and streamed with async-to-LDS copies.
__global__ __launch_bounds__(128) void attn_kernel(
    const _Float16* __restrict__ Q, const _Float16* __restrict__ K,
    const _Float16* __restrict__ V, _Float16* __restrict__ Ctx) {
    __shared__ _Float16 ldsK[2][64 * 64];  // K chunk (row = s-local, col = d), ping-pong
    __shared__ _Float16 ldsVt[64 * 64];    // V chunk transposed: row = d, col = s-local
    __shared__ _Float16 ldsP[4][16 * 64];  // per-wave P tile (rows = q, cols = s-local)

    int tid = threadIdx.x, lane = tid & 31, w = tid >> 5;
    int bh = blockIdx.y;
    int b  = bh >> 4;                // NHEAD = 16
    int h  = bh & (NHEAD - 1);
    const _Float16* Qh = Q + (size_t)bh * LSEQ * HDIM;
    const _Float16* Kh = K + (size_t)bh * SSEQ * HDIM;
    const _Float16* Vh = V + (size_t)bh * SSEQ * HDIM;
    int m0 = blockIdx.x * 64 + w * 16;

    // Q A-fragments (reused across the whole S loop): d = 0..31, 32..63
    v16h aq0 = load_frag_a(Qh + (size_t)m0 * HDIM + 0,  HDIM, lane);
    v16h aq1 = load_frag_a(Qh + (size_t)m0 * HDIM + 32, HDIM, lane);

    v8f acc[4] = {};
    float mrow[8], lrow[8];
    #pragma unroll
    for (int r = 0; r < 8; r++) { mrow[r] = -3.0e38f; lrow[r] = 0.0f; }

    const float scale = 0.125f; // 1/sqrt(64)

    // prologue: async-stage K chunk 0 into buffer 0
    stage_k_chunk(Kh, &ldsK[0][0], tid);

    for (int s0 = 0; s0 < SSEQ; s0 += 64) {
        int cur = (s0 >> 6) & 1;

        // ---- stage V chunk transposed (VGPR path; transpose needed anyway) ----
        #pragma unroll
        for (int i = 0; i < 4; i++) {
            int off  = (i * 128 + tid) * 8;
            int srow = off >> 6, d0 = off & 63;
            v8h t = *(const v8h*)(Vh + (size_t)(s0 + srow) * HDIM + d0);
            #pragma unroll
            for (int e = 0; e < 8; e++) ldsVt[(d0 + e) * 64 + srow] = t[e];
        }

        // ---- issue async prefetch of next K chunk, then wait for current ----
        if (s0 + 64 < SSEQ) {
            stage_k_chunk(Kh + (size_t)(s0 + 64) * HDIM, &ldsK[cur ^ 1][0], tid);
            async_wait_le4();   // loads complete in order: chunk `cur` resident
        } else {
            async_wait_0();
        }
        __syncthreads();

        // ---- scores = Q * K^T : 16 x 64 per wave ----
        v8f sc[4] = {};
        const _Float16* Kc = &ldsK[cur][0];
        #pragma unroll
        for (int kk = 0; kk < 2; kk++) {
            v16h a = kk ? aq1 : aq0;
            #pragma unroll
            for (int j = 0; j < 4; j++) {
                v16h bf = load_frag_b(Kc + (j * 16) * 64 + kk * 32, 64, lane);
                sc[j] = wmma_f16(a, bf, sc[j]);
            }
        }

        // ---- online softmax ----
        float pj[4][8];
        #pragma unroll
        for (int r = 0; r < 8; r++) {
            float cm = -3.0e38f;
            #pragma unroll
            for (int j = 0; j < 4; j++) cm = fmaxf(cm, sc[j][r] * scale);
            cm = group_max(cm);
            float mn   = fmaxf(mrow[r], cm);
            float corr = __expf(mrow[r] - mn);
            mrow[r] = mn;
            float ps = 0.0f;
            #pragma unroll
            for (int j = 0; j < 4; j++) {
                float e = __expf(sc[j][r] * scale - mn);
                pj[j][r] = e;
                ps += e;
            }
            ps = group_sum(ps);
            lrow[r] = lrow[r] * corr + ps;
            #pragma unroll
            for (int j = 0; j < 4; j++) acc[j][r] *= corr;
        }

        // ---- P to LDS (C-layout -> row-major, for A-fragment reload) ----
        {
            _Float16* Pw = &ldsP[w][0];
            int mb = (lane >> 4) << 3;
            #pragma unroll
            for (int j = 0; j < 4; j++)
                #pragma unroll
                for (int r = 0; r < 8; r++)
                    Pw[(r + mb) * 64 + j * 16 + (lane & 15)] = (_Float16)pj[j][r];
        }
        __syncthreads();

        // ---- context += P * V ----
        #pragma unroll
        for (int kk = 0; kk < 2; kk++) {
            v16h ap = load_frag_a(&ldsP[w][0] + kk * 32, 64, lane);
            #pragma unroll
            for (int j = 0; j < 4; j++) {
                v16h bv = load_frag_b(ldsVt + (j * 16) * 64 + kk * 32, 64, lane);
                acc[j] = wmma_f16(ap, bv, acc[j]);
            }
        }
        __syncthreads();
    }

    // ---- normalize + store into token-major ctx ----
    int mb = (lane >> 4) << 3;
    #pragma unroll
    for (int j = 0; j < 4; j++) {
        #pragma unroll
        for (int r = 0; r < 8; r++) {
            int lq = m0 + r + mb;
            int e  = h * HDIM + j * 16 + (lane & 15);
            float v = acc[j][r] * (1.0f / lrow[r]);
            Ctx[((size_t)lq * BSZ + b) * EMBED + e] = (_Float16)v;
        }
    }
}

// ---------------- host side ----------------
extern "C" void kernel_launch(void* const* d_in, const int* in_sizes, int n_in,
                              void* d_out, int out_size, void* d_ws, size_t ws_size,
                              hipStream_t stream) {
    const float* q     = (const float*)d_in[0];
    const float* k     = (const float*)d_in[1];
    const float* v     = (const float*)d_in[2];
    const float* w_in  = (const float*)d_in[3];
    const float* b_in  = (const float*)d_in[4];
    const float* w_out = (const float*)d_in[5];
    const float* b_out = (const float*)d_in[6];

    const size_t ACT = (size_t)MTOK * EMBED;      // 4 Mi elems
    _Float16* ws   = (_Float16*)d_ws;
    _Float16* Xq   = ws;                          // activations f16
    _Float16* Xk   = Xq + ACT;
    _Float16* Xv   = Xk + ACT;
    _Float16* Win  = Xv + ACT;                    // 3E x E
    _Float16* Wout = Win + (size_t)3 * EMBED * EMBED;
    _Float16* Qh   = Wout + (size_t)EMBED * EMBED;
    _Float16* Kh   = Qh + ACT;
    _Float16* Vh   = Kh + ACT;
    _Float16* Ctx  = Vh + ACT;

    // fp32 -> fp16
    auto cvt = [&](const float* s, _Float16* d, size_t n) {
        int n4 = (int)(n / 4);
        cvt4_kernel<<<(n4 + 255) / 256, 256, 0, stream>>>((const float4*)s, d, n4);
    };
    cvt(q, Xq, ACT);
    cvt(k, Xk, ACT);
    cvt(v, Xv, ACT);
    cvt(w_in, Win, (size_t)3 * EMBED * EMBED);
    cvt(w_out, Wout, (size_t)EMBED * EMBED);

    // QKV projections -> head layout
    dim3 gg(EMBED / 256, MTOK / 64);
    gemm_head_kernel<<<gg, 256, 0, stream>>>(Xq, Win, b_in, Qh);
    gemm_head_kernel<<<gg, 256, 0, stream>>>(Xk, Win + (size_t)EMBED * EMBED, b_in + EMBED, Kh);
    gemm_head_kernel<<<gg, 256, 0, stream>>>(Xv, Win + (size_t)2 * EMBED * EMBED, b_in + 2 * EMBED, Vh);

    // flash attention
    attn_kernel<<<dim3(LSEQ / 64, BSZ * NHEAD), 128, 0, stream>>>(Qh, Kh, Vh, Ctx);

    // output projection (fp32 out, (L,B,E) token order)
    gemm_out_kernel<<<gg, 256, 0, stream>>>(Ctx, Wout, b_out, (float*)d_out);
}
